// GraphAttentionLayer_82454782149144
// MI455X (gfx1250) — matrix operand order repaired and checked by reading, not compiled, not run on hardware
//
#include <hip/hip_runtime.h>
#include <hip/hip_bf16.h>

#define N_NODES 2048
#define IN_F    128
#define OUT_F   64
#define LEAKY   0.01f

typedef __bf16 v16bf __attribute__((ext_vector_type(16)));
typedef __bf16 v8bf  __attribute__((ext_vector_type(8)));
typedef float  v8f   __attribute__((ext_vector_type(8)));

static __device__ __forceinline__ v16bf cat8(v8bf lo, v8bf hi) {
  return __builtin_shufflevector(lo, hi, 0,1,2,3,4,5,6,7,8,9,10,11,12,13,14,15);
}

// ---------------------------------------------------------------------------
// Kernel 1: Wh = h @ W (bf16 WMMA, f32 accum), store Wh^T bf16 for pass 2,
// plus rank-1 scores s1 = Wh@a1, s2 = Wh@a2 (f32).
// ---------------------------------------------------------------------------
__global__ void __launch_bounds__(128)
gat_precompute(const float* __restrict__ h, const float* __restrict__ W,
               const float* __restrict__ a, __bf16* __restrict__ WhT,
               float* __restrict__ s1g, float* __restrict__ s2g) {
  __shared__ __bf16 hbf[16 * IN_F];        // h tile, bf16, row-major
  __shared__ __bf16 wtb[OUT_F * IN_F];     // W^T, bf16: [c][k]
  __shared__ float  whf[16 * OUT_F];       // Wh tile, f32

  const int tid = threadIdx.x;
  const int b   = blockIdx.x >> 7;         // 128 row-tiles per batch
  const int i0  = (blockIdx.x & 127) << 4;

  #pragma unroll
  for (int r = 0; r < 16; ++r)
    hbf[r * IN_F + tid] = (__bf16)h[((size_t)b * N_NODES + i0 + r) * IN_F + tid];
  for (int e = tid; e < IN_F * OUT_F; e += 128) {
    int k = e >> 6, c = e & 63;            // W is [128][64] row-major
    wtb[c * IN_F + k] = (__bf16)W[e];
  }
  __syncthreads();

  const int lane = tid & 31, l = lane & 15, hh = lane >> 4;
  const int c0 = (tid >> 5) << 4;          // wave's 16-column chunk

  v8f acc = {};
  #pragma unroll
  for (int k0 = 0; k0 < IN_F; k0 += 32) {
    v8bf alo = *(const v8bf*)(hbf + l * IN_F + k0 + 8 * hh);
    v8bf ahi = *(const v8bf*)(hbf + l * IN_F + k0 + 16 + 8 * hh);
    v16bf afrag = cat8(alo, ahi);
    v16bf bfrag = cat8(*(const v8bf*)(wtb + (c0 + l) * IN_F + k0 + 8 * hh),
                       *(const v8bf*)(wtb + (c0 + l) * IN_F + k0 + 16 + 8 * hh));
    acc = __builtin_amdgcn_wmma_f32_16x16x32_bf16(false, afrag, false, bfrag,
                                                  (short)0, acc, false, false);
  }
  #pragma unroll
  for (int v = 0; v < 8; ++v)
    whf[(v + 8 * hh) * OUT_F + c0 + l] = acc[v];
  __syncthreads();

  if (tid < 16) {
    float d1 = 0.f, d2 = 0.f;
    #pragma unroll 8
    for (int c = 0; c < OUT_F; ++c) {
      float w = whf[tid * OUT_F + c];
      d1 += w * a[c];
      d2 += w * a[OUT_F + c];
    }
    s1g[(size_t)b * N_NODES + i0 + tid] = d1;
    s2g[(size_t)b * N_NODES + i0 + tid] = d2;
  }
  for (int e = tid; e < 16 * OUT_F; e += 128) {
    int r = e >> 6, c = e & 63;
    WhT[((size_t)b * OUT_F + c) * N_NODES + i0 + r] = (__bf16)whf[e];
  }
}

// ---------------------------------------------------------------------------
// Kernel 2: fused masked softmax + attn@Wh + elu.
// 8 threads per row x 16 rows. Pass A: masked min/max of s2 per row (rowmax
// via leakyrelu monotonicity; compares only, no exp). Pass B: unnormalized
// P-tiles (exp once per element) fed to bf16 WMMA, with on-the-fly row sums;
// normalization deferred to the epilogue. Double-buffered P => 1 barrier/iter.
// ---------------------------------------------------------------------------
__global__ void __launch_bounds__(128)
gat_attention(const int* __restrict__ adj, const __bf16* __restrict__ WhT,
              const float* __restrict__ s1g, const float* __restrict__ s2g,
              float* __restrict__ outp) {
  __shared__ float  s2_sh[N_NODES];                 // 8 KB
  __shared__ float  s1_sh[16], rowmax_sh[16], rowsum_sh[16];
  __shared__ __bf16 Plds[2][16 * 64];               // double-buffered P tile

  const int tid = threadIdx.x;
  const int b   = blockIdx.x >> 7;
  const int i0  = (blockIdx.x & 127) << 4;
  const int r   = tid >> 3;                          // row 0..15
  const int g   = tid & 7;                           // 8 threads per row

  for (int e = tid; e < N_NODES; e += 128)
    s2_sh[e] = s2g[(size_t)b * N_NODES + e];
  if (tid < 16) s1_sh[tid] = s1g[(size_t)b * N_NODES + i0 + tid];
  __syncthreads();

  const int4* __restrict__ arow4 =
      (const int4*)(adj + ((size_t)b * N_NODES + i0 + r) * (size_t)N_NODES);

  // ---- Pass A: masked min/max of s2 per row (single HBM sweep of adj) ----
  float mx = -1e30f, mn = 1e30f;
  for (int k = g; k < N_NODES / 4; k += 8) {
    __builtin_prefetch((const char*)(arow4 + k) + 4096, 0, 0);
    int4 av = arow4[k];
    int jb = k * 4;
    if (av.x > 0) { float v = s2_sh[jb + 0]; mx = fmaxf(mx, v); mn = fminf(mn, v); }
    if (av.y > 0) { float v = s2_sh[jb + 1]; mx = fmaxf(mx, v); mn = fminf(mn, v); }
    if (av.z > 0) { float v = s2_sh[jb + 2]; mx = fmaxf(mx, v); mn = fminf(mn, v); }
    if (av.w > 0) { float v = s2_sh[jb + 3]; mx = fmaxf(mx, v); mn = fminf(mn, v); }
  }
  #pragma unroll
  for (int m = 1; m < 8; m <<= 1) {
    mx = fmaxf(mx, __shfl_xor(mx, m, 8));
    mn = fminf(mn, __shfl_xor(mn, m, 8));
  }
  if (g == 0) {
    float s1v = s1_sh[r];
    float x = s1v * (s1v >= 0.f ? mx : mn);   // LR is monotone increasing
    rowmax_sh[r] = x > 0.f ? x : LEAKY * x;
  }
  __syncthreads();

  // ---- Pass B: P-tiles + WMMA + on-the-fly row sums ----
  const int lane = tid & 31, l = lane & 15, hh = lane >> 4;
  const int c0 = (tid >> 5) << 4;
  const __bf16* __restrict__ whtB =
      WhT + ((size_t)b * OUT_F + c0 + l) * (size_t)N_NODES;
  const float s1v  = s1_sh[r];
  const float rmax = rowmax_sh[r];

  auto pcalc = [&](int m, int j) -> float {
    if (m <= 0) return 0.f;
    float t = s1v * s2_sh[j];
    float e = t > 0.f ? t : LEAKY * t;
    return __expf(e - rmax);                  // <= 1 by construction
  };

  v8f acc = {};
  float psum = 0.f;
  int buf = 0;
  for (int j0 = 0; j0 < N_NODES; j0 += 64, buf ^= 1) {
    const int kb = (j0 >> 2) + g * 2;
    int4 a0 = arow4[kb];
    int4 a1 = arow4[kb + 1];
    const int jb = j0 + g * 8;
    float p0 = pcalc(a0.x, jb + 0), p1 = pcalc(a0.y, jb + 1);
    float p2 = pcalc(a0.z, jb + 2), p3 = pcalc(a0.w, jb + 3);
    float p4 = pcalc(a1.x, jb + 4), p5 = pcalc(a1.y, jb + 5);
    float p6 = pcalc(a1.z, jb + 6), p7 = pcalc(a1.w, jb + 7);
    psum += ((p0 + p1) + (p2 + p3)) + ((p4 + p5) + (p6 + p7));
    v8bf pv;
    pv[0] = (__bf16)p0; pv[1] = (__bf16)p1; pv[2] = (__bf16)p2; pv[3] = (__bf16)p3;
    pv[4] = (__bf16)p4; pv[5] = (__bf16)p5; pv[6] = (__bf16)p6; pv[7] = (__bf16)p7;
    *(v8bf*)(&Plds[buf][r * 64 + g * 8]) = pv;
    __syncthreads();

    const __bf16* PL = &Plds[buf][0];
    v16bf af0 = cat8(*(const v8bf*)(PL + l * 64 + 8 * hh),
                     *(const v8bf*)(PL + l * 64 + 16 + 8 * hh));
    v16bf af1 = cat8(*(const v8bf*)(PL + l * 64 + 32 + 8 * hh),
                     *(const v8bf*)(PL + l * 64 + 48 + 8 * hh));
    v16bf bf0 = cat8(*(const v8bf*)(whtB + j0 + 8 * hh),
                     *(const v8bf*)(whtB + j0 + 16 + 8 * hh));
    v16bf bf1 = cat8(*(const v8bf*)(whtB + j0 + 32 + 8 * hh),
                     *(const v8bf*)(whtB + j0 + 48 + 8 * hh));
    acc = __builtin_amdgcn_wmma_f32_16x16x32_bf16(false, af0, false, bf0,
                                                  (short)0, acc, false, false);
    acc = __builtin_amdgcn_wmma_f32_16x16x32_bf16(false, af1, false, bf1,
                                                  (short)0, acc, false, false);
  }

  // ---- Row-sum reduction (deterministic shuffle tree over 8 lanes) ----
  #pragma unroll
  for (int m = 1; m < 8; m <<= 1) psum += __shfl_xor(psum, m, 8);
  if (g == 0) rowsum_sh[r] = psum;
  __syncthreads();

  // ---- Epilogue: normalize, elu, store ----
  #pragma unroll
  for (int v = 0; v < 8; ++v) {
    int rr = v + 8 * hh;
    float S = rowsum_sh[rr];
    float x = S > 0.f ? acc[v] / S : 0.f;     // empty row -> 0 -> elu(0)=0
    float y = x > 0.f ? x : (__expf(x) - 1.f);
    outp[((size_t)b * N_NODES + i0 + rr) * OUT_F + c0 + l] = y;
  }
}

// ---------------------------------------------------------------------------
extern "C" void kernel_launch(void* const* d_in, const int* in_sizes, int n_in,
                              void* d_out, int out_size, void* d_ws, size_t ws_size,
                              hipStream_t stream) {
  (void)in_sizes; (void)n_in; (void)out_size; (void)ws_size;
  const float* h   = (const float*)d_in[0];
  const int*   adj = (const int*)d_in[1];
  const float* W   = (const float*)d_in[2];
  const float* a   = (const float*)d_in[3];
  float* out = (float*)d_out;

  char* ws = (char*)d_ws;
  const size_t whtBytes = (size_t)8 * OUT_F * N_NODES * sizeof(__bf16); // 2 MB
  __bf16* WhT = (__bf16*)ws;
  float*  s1  = (float*)(ws + whtBytes);
  float*  s2  = s1 + (size_t)8 * N_NODES;

  const int blocks = 8 * (N_NODES / 16);   // 1024
  gat_precompute<<<blocks, 128, 0, stream>>>(h, W, a, WhT, s1, s2);
  gat_attention <<<blocks, 128, 0, stream>>>(adj, WhT, s1, s2, out);
}
